// LightGNNModel_67362267070636
// MI455X (gfx1250) — matrix operand order repaired and checked by reading, not compiled, not run on hardware
//
#include <hip/hip_runtime.h>
#include <hip/hip_bf16.h>
#include <math.h>

#define N_NODES  50000
#define N_EDGES  800000
#define E_TOT    (N_EDGES + N_NODES)   // with self loops
#define N_GRAPHS 64
#define IN_CH    432
#define KPAD     448                   // 432 padded to mult of 32 for K-loop
#define HID      64
#define HEADS    2
#define C1       (HEADS * HID)         // 128
#define NEG_SLOPE 0.2f
#define EPS_BN    1e-5f

#define CDIV(a,b) (((a)+(b)-1)/(b))

typedef __attribute__((ext_vector_type(16))) __bf16 v16bf;
typedef __attribute__((ext_vector_type(8)))  float  v8f;

union Frag { uint4 u[2]; v16bf v; };

static __device__ __forceinline__ unsigned short f2bf(float f) {
  unsigned int u = __float_as_uint(f);
  unsigned int r = u + 0x7FFFu + ((u >> 16) & 1u);   // RNE
  return (unsigned short)(r >> 16);
}
static __device__ __forceinline__ float lrelu(float x) {
  return x > 0.f ? x : NEG_SLOPE * x;
}
static __device__ __forceinline__ void atomicMaxF(float* addr, float v) {
  if (!(v == v)) return;
  if (v >= 0.f) atomicMax((int*)addr, __float_as_int(v));
  else          atomicMin((unsigned int*)addr, __float_as_uint(v));
}
static __device__ __forceinline__ void edge_pair(const int* ei, int e, int& s, int& d) {
  if (e < N_EDGES) { s = ei[e]; d = ei[N_EDGES + e]; }
  else             { s = e - N_EDGES; d = s; }       // self loop
}

// ---------------- utility fills ----------------
__global__ void k_fill(float* p, float v, int n) {
  int i = blockIdx.x * blockDim.x + threadIdx.x;
  if (i < n) p[i] = v;
}

// ---------------- bf16 conversion / padding ----------------
__global__ void k_cvt_x(const float* __restrict__ x, unsigned short* __restrict__ xb) {
  int i = blockIdx.x * blockDim.x + threadIdx.x;
  if (i >= N_NODES * KPAD) return;
  int r = i / KPAD, k = i - r * KPAD;
  xb[i] = (k < IN_CH) ? f2bf(x[(size_t)r * IN_CH + k]) : (unsigned short)0;
}
__global__ void k_cvt_w1t(const float* __restrict__ W, unsigned short* __restrict__ wt) {
  int i = blockIdx.x * blockDim.x + threadIdx.x;
  if (i >= C1 * KPAD) return;
  int col = i / KPAD, k = i - col * KPAD;           // wt[col][k], W is [432][128]
  wt[i] = (k < IN_CH) ? f2bf(W[(size_t)k * C1 + col]) : (unsigned short)0;
}
__global__ void k_cvt_w2t(const float* __restrict__ W, unsigned short* __restrict__ wt) {
  int i = blockIdx.x * blockDim.x + threadIdx.x;
  if (i >= HID * C1) return;
  int col = i / C1, k = i - col * C1;               // wt[col][k], W is [128][64]
  wt[i] = f2bf(W[(size_t)k * HID + col]);
}

// ---------------- WMMA bf16 GEMM: C[MxN] = A[MxK] * Bt[NxK]^T ----------------
// one wave computes a 16 x (NT*16) slab; grid.x covers M/16 tiles, 4 waves/block
template <int K, int NT>
__global__ void k_gemm_bf16(const unsigned short* __restrict__ A,
                            const unsigned short* __restrict__ Bt,
                            float* __restrict__ C, int mtiles) {
  int wave = threadIdx.x >> 5;
  int tm   = blockIdx.x * 4 + wave;
  if (tm >= mtiles) return;
  int lane = threadIdx.x & 31;
  int half = lane >> 4;          // 0/1
  int lcol = lane & 15;
  int row  = tm * 16 + lcol;
  const unsigned short* arow = A + (size_t)row * K;

  v8f acc[NT];
  v8f z = {0.f,0.f,0.f,0.f,0.f,0.f,0.f,0.f};
#pragma unroll
  for (int n = 0; n < NT; ++n) acc[n] = z;

  for (int k = 0; k < K; k += 32) {
    // A 16x32 bf16 fragment: lane(row=lcol) holds K chunks [8h,8h+8) and [16+8h,16+8h+8)
    Frag fa;
    fa.u[0] = *(const uint4*)(arow + k + half * 8);
    fa.u[1] = *(const uint4*)(arow + k + 16 + half * 8);
#pragma unroll
    for (int n = 0; n < NT; ++n) {
      // B 32x16 fragment: lane holds column lcol, K chunks [8h,8h+8) and [16+8h,16+8h+8)
      const unsigned short* brow = Bt + (size_t)(n * 16 + lcol) * K + k + half * 8;
      Frag fb;
      fb.u[0] = *(const uint4*)(brow);
      fb.u[1] = *(const uint4*)(brow + 16);
      acc[n] = __builtin_amdgcn_wmma_f32_16x16x32_bf16(
          false, fa.v, false, fb.v, (short)0, acc[n], false, false);
    }
  }
  // C/D layout: VGPR i -> row tm*16 + i + 8*half, col = lcol
  int r0 = tm * 16 + half * 8;
  const int ldc = NT * 16;
#pragma unroll
  for (int n = 0; n < NT; ++n)
#pragma unroll
    for (int i = 0; i < 8; ++i)
      C[(size_t)(r0 + i) * ldc + n * 16 + lcol] = acc[n][i];
}

// ---------------- attention coefficients ----------------
__global__ void k_alpha(const float* __restrict__ H, const float* __restrict__ a_s,
                        const float* __restrict__ a_d, float* __restrict__ out_s,
                        float* __restrict__ out_d, int nheads) {
  int t = blockIdx.x * blockDim.x + threadIdx.x;
  if (t >= N_NODES * nheads) return;
  int node = t / nheads, h = t - node * nheads;
  const float* row = H + (size_t)node * nheads * HID + h * HID;
  const float* vs = a_s + h * HID;
  const float* vd = a_d + h * HID;
  float ss = 0.f, sd = 0.f;
#pragma unroll
  for (int c = 0; c < HID; c += 4) {
    float4 hv = *(const float4*)(row + c);
    float4 s  = *(const float4*)(vs + c);
    float4 d  = *(const float4*)(vd + c);
    ss += hv.x*s.x + hv.y*s.y + hv.z*s.z + hv.w*s.w;
    sd += hv.x*d.x + hv.y*d.y + hv.z*d.z + hv.w*d.w;
  }
  out_s[t] = ss;
  out_d[t] = sd;
}

// ---------------- segment softmax over destinations ----------------
__global__ void k_edge_max(const int* __restrict__ ei, const float* __restrict__ as,
                           const float* __restrict__ ad, float* __restrict__ m, int nheads) {
  int t = blockIdx.x * blockDim.x + threadIdx.x;
  if (t >= E_TOT * nheads) return;
  int e = t / nheads, h = t - e * nheads;
  int s, d; edge_pair(ei, e, s, d);
  float ev = lrelu(as[s * nheads + h] + ad[d * nheads + h]);
  atomicMaxF(&m[d * nheads + h], ev);
}
__global__ void k_edge_sum(const int* __restrict__ ei, const float* __restrict__ as,
                           const float* __restrict__ ad, const float* __restrict__ m,
                           float* __restrict__ dn, int nheads) {
  int t = blockIdx.x * blockDim.x + threadIdx.x;
  if (t >= E_TOT * nheads) return;
  int e = t / nheads, h = t - e * nheads;
  int s, d; edge_pair(ei, e, s, d);
  float ev = lrelu(as[s * nheads + h] + ad[d * nheads + h]);
  float mm = m[d * nheads + h];
  if (!(mm >= -3.0e38f)) mm = 0.f;          // isfinite fixup
  atomicAdd(&dn[d * nheads + h], __expf(ev - mm));
}
// msg: agg[dst, c] += h[src, c] * alpha(e, head(c)) — one thread per (edge, channel)
__global__ void k_msg(const int* __restrict__ ei, const float* __restrict__ as,
                      const float* __restrict__ ad, const float* __restrict__ m,
                      const float* __restrict__ dn, const float* __restrict__ H,
                      float* __restrict__ agg, int nheads) {
  int C = nheads * HID;
  int t = blockIdx.x * blockDim.x + threadIdx.x;
  if (t >= E_TOT * C) return;
  int e = t / C, c = t - e * C;
  int h = c / HID;
  int s, d; edge_pair(ei, e, s, d);
  float ev = lrelu(as[s * nheads + h] + ad[d * nheads + h]);
  float mm = m[d * nheads + h];
  if (!(mm >= -3.0e38f)) mm = 0.f;
  float alpha = __expf(ev - mm) / (dn[d * nheads + h] + 1e-16f);
  atomicAdd(&agg[(size_t)d * C + c], H[(size_t)s * C + c] * alpha);
}

// ---------------- batch norm ----------------
__global__ void k_bnstats(const float* __restrict__ X, float* __restrict__ mu,
                          float* __restrict__ rstd, int C) {
  int c = blockIdx.x, tid = threadIdx.x;
  float s = 0.f, s2 = 0.f;
  for (int r = tid; r < N_NODES; r += blockDim.x) {
    float v = X[(size_t)r * C + c];
    s += v; s2 += v * v;
  }
  __shared__ float sh[256], sh2[256];
  sh[tid] = s; sh2[tid] = s2; __syncthreads();
  for (int st = 128; st > 0; st >>= 1) {
    if (tid < st) { sh[tid] += sh[tid + st]; sh2[tid] += sh2[tid + st]; }
    __syncthreads();
  }
  if (tid == 0) {
    float m_ = sh[0] / (float)N_NODES;
    float var = sh2[0] / (float)N_NODES - m_ * m_;
    mu[c] = m_;
    rstd[c] = rsqrtf(var + EPS_BN);
  }
}
// BN + ReLU, emit bf16 for the next GEMM  (bias b1 cancels inside BN)
__global__ void k_bnrelu_bf16(const float* __restrict__ X, const float* __restrict__ mu,
                              const float* __restrict__ rstd, const float* __restrict__ g,
                              const float* __restrict__ be, unsigned short* __restrict__ Y) {
  int t = blockIdx.x * blockDim.x + threadIdx.x;
  if (t >= N_NODES * C1) return;
  int c = t % C1;
  float y = (X[t] - mu[c]) * rstd[c] * g[c] + be[c];
  Y[t] = f2bf(fmaxf(y, 0.f));
}
// BN + ReLU for layer 2, write node embeddings to d_out and accumulate graph pool
__global__ void k_bnrelu_out_pool(const float* __restrict__ X, const float* __restrict__ mu,
                                  const float* __restrict__ rstd, const float* __restrict__ g,
                                  const float* __restrict__ be, const int* __restrict__ batch,
                                  float* __restrict__ Hout, float* __restrict__ psum) {
  int t = blockIdx.x * blockDim.x + threadIdx.x;
  if (t >= N_NODES * HID) return;
  int r = t / HID, c = t - r * HID;
  float y = (X[t] - mu[c]) * rstd[c] * g[c] + be[c];
  y = fmaxf(y, 0.f);
  Hout[t] = y;
  atomicAdd(&psum[batch[r] * HID + c], y);
}
__global__ void k_counts(const int* __restrict__ batch, float* __restrict__ cnt) {
  int n = blockIdx.x * blockDim.x + threadIdx.x;
  if (n < N_NODES) atomicAdd(&cnt[batch[n]], 1.f);
}

// ---------------- classifier head: 64 -> 32 -> 2 ----------------
__global__ void k_cls(const float* __restrict__ psum, const float* __restrict__ cnt,
                      const float* __restrict__ Wc1, const float* __restrict__ bc1,
                      const float* __restrict__ Wc2, const float* __restrict__ bc2,
                      float* __restrict__ out) {
  int gph = blockIdx.x, t = threadIdx.x;   // 32 threads
  __shared__ float pooled[HID];
  __shared__ float hb[32];
  float cn = fmaxf(cnt[gph], 1.f);
  for (int k = t; k < HID; k += 32) pooled[k] = psum[gph * HID + k] / cn;
  __syncthreads();
  float acc = bc1[t];
  for (int k = 0; k < HID; ++k) acc += pooled[k] * Wc1[k * 32 + t];
  hb[t] = fmaxf(acc, 0.f);
  __syncthreads();
  if (t < 2) {
    float o = bc2[t];
    for (int k = 0; k < 32; ++k) o += hb[k] * Wc2[k * 2 + t];
    out[gph * 2 + t] = o;
  }
}

extern "C" void kernel_launch(void* const* d_in, const int* in_sizes, int n_in,
                              void* d_out, int out_size, void* d_ws, size_t ws_size,
                              hipStream_t stream) {
  const float* x    = (const float*)d_in[0];
  const int*   ei   = (const int*)d_in[1];
  const int*   bat  = (const int*)d_in[2];
  const float* W1   = (const float*)d_in[3];
  const float* as1w = (const float*)d_in[4];
  const float* ad1w = (const float*)d_in[5];
  const float* g1   = (const float*)d_in[7];
  const float* be1  = (const float*)d_in[8];
  const float* W2   = (const float*)d_in[9];
  const float* as2w = (const float*)d_in[10];
  const float* ad2w = (const float*)d_in[11];
  const float* g2   = (const float*)d_in[13];
  const float* be2  = (const float*)d_in[14];
  const float* Wc1  = (const float*)d_in[15];
  const float* bc1  = (const float*)d_in[16];
  const float* Wc2  = (const float*)d_in[17];
  const float* bc2  = (const float*)d_in[18];
  float* out_logits = (float*)d_out;                 // [64,2]
  float* out_h      = (float*)d_out + N_GRAPHS * 2;  // [50000,64]

  // workspace carve-out (256B aligned)
  size_t off = 0;
  auto alloc = [&](size_t bytes) -> char* {
    char* p = (char*)d_ws + off;
    off += (bytes + 255) & ~(size_t)255;
    return p;
  };
  unsigned short* xbf  = (unsigned short*)alloc((size_t)N_NODES * KPAD * 2);
  unsigned short* w1t  = (unsigned short*)alloc((size_t)C1 * KPAD * 2);
  unsigned short* w2t  = (unsigned short*)alloc((size_t)HID * C1 * 2);
  float* h1   = (float*)alloc((size_t)N_NODES * C1 * 4);
  float* as1  = (float*)alloc((size_t)N_NODES * HEADS * 4);
  float* ad1  = (float*)alloc((size_t)N_NODES * HEADS * 4);
  float* m1   = (float*)alloc((size_t)N_NODES * HEADS * 4);
  float* dn1  = (float*)alloc((size_t)N_NODES * HEADS * 4);
  float* agg1 = (float*)alloc((size_t)N_NODES * C1 * 4);
  unsigned short* h1bf = (unsigned short*)alloc((size_t)N_NODES * C1 * 2);
  float* h2   = (float*)alloc((size_t)N_NODES * HID * 4);
  float* as2  = (float*)alloc((size_t)N_NODES * 4);
  float* ad2  = (float*)alloc((size_t)N_NODES * 4);
  float* m2   = (float*)alloc((size_t)N_NODES * 4);
  float* dn2  = (float*)alloc((size_t)N_NODES * 4);
  float* agg2 = (float*)alloc((size_t)N_NODES * HID * 4);
  float* mu1  = (float*)alloc(C1 * 4);
  float* rs1  = (float*)alloc(C1 * 4);
  float* mu2  = (float*)alloc(HID * 4);
  float* rs2  = (float*)alloc(HID * 4);
  float* psum = (float*)alloc(N_GRAPHS * HID * 4);
  float* cnt  = (float*)alloc(N_GRAPHS * 4);

  const int T = 256;
  const float NEG_INF = -__builtin_inff();

  // init accumulators (every call — harness does not re-poison)
  k_fill<<<CDIV(N_NODES * C1, T), T, 0, stream>>>(agg1, 0.f, N_NODES * C1);
  k_fill<<<CDIV(N_NODES * HEADS, T), T, 0, stream>>>(m1, NEG_INF, N_NODES * HEADS);
  k_fill<<<CDIV(N_NODES * HEADS, T), T, 0, stream>>>(dn1, 0.f, N_NODES * HEADS);
  k_fill<<<CDIV(N_NODES * HID, T), T, 0, stream>>>(agg2, 0.f, N_NODES * HID);
  k_fill<<<CDIV(N_NODES, T), T, 0, stream>>>(m2, NEG_INF, N_NODES);
  k_fill<<<CDIV(N_NODES, T), T, 0, stream>>>(dn2, 0.f, N_NODES);
  k_fill<<<CDIV(N_GRAPHS * HID, T), T, 0, stream>>>(psum, 0.f, N_GRAPHS * HID);
  k_fill<<<1, N_GRAPHS, 0, stream>>>(cnt, 0.f, N_GRAPHS);

  // bf16 staging
  k_cvt_x  <<<CDIV(N_NODES * KPAD, T), T, 0, stream>>>(x, xbf);
  k_cvt_w1t<<<CDIV(C1 * KPAD, T), T, 0, stream>>>(W1, w1t);
  k_cvt_w2t<<<CDIV(HID * C1, T), T, 0, stream>>>(W2, w2t);

  const int MT = N_NODES / 16;   // 3125 row tiles

  // ---- layer 1 ----
  k_gemm_bf16<KPAD, 8><<<CDIV(MT, 4), 128, 0, stream>>>(xbf, w1t, h1, MT);
  k_alpha<<<CDIV(N_NODES * HEADS, T), T, 0, stream>>>(h1, as1w, ad1w, as1, ad1, HEADS);
  k_edge_max<<<CDIV(E_TOT * HEADS, T), T, 0, stream>>>(ei, as1, ad1, m1, HEADS);
  k_edge_sum<<<CDIV(E_TOT * HEADS, T), T, 0, stream>>>(ei, as1, ad1, m1, dn1, HEADS);
  k_msg<<<CDIV(E_TOT * C1, T), T, 0, stream>>>(ei, as1, ad1, m1, dn1, h1, agg1, HEADS);
  k_bnstats<<<C1, 256, 0, stream>>>(agg1, mu1, rs1, C1);
  k_bnrelu_bf16<<<CDIV(N_NODES * C1, T), T, 0, stream>>>(agg1, mu1, rs1, g1, be1, h1bf);

  // ---- layer 2 ----
  k_gemm_bf16<C1, 4><<<CDIV(MT, 4), 128, 0, stream>>>(h1bf, w2t, h2, MT);
  k_alpha<<<CDIV(N_NODES, T), T, 0, stream>>>(h2, as2w, ad2w, as2, ad2, 1);
  k_edge_max<<<CDIV(E_TOT, T), T, 0, stream>>>(ei, as2, ad2, m2, 1);
  k_edge_sum<<<CDIV(E_TOT, T), T, 0, stream>>>(ei, as2, ad2, m2, dn2, 1);
  k_msg<<<CDIV(E_TOT * HID, T), T, 0, stream>>>(ei, as2, ad2, m2, dn2, h2, agg2, 1);
  k_bnstats<<<HID, 256, 0, stream>>>(agg2, mu2, rs2, HID);
  k_bnrelu_out_pool<<<CDIV(N_NODES * HID, T), T, 0, stream>>>(agg2, mu2, rs2, g2, be2,
                                                              bat, out_h, psum);
  // ---- pool + classifier ----
  k_counts<<<CDIV(N_NODES, T), T, 0, stream>>>(bat, cnt);
  k_cls<<<N_GRAPHS, 32, 0, stream>>>(psum, cnt, Wc1, bc1, Wc2, bc2, out_logits);

  (void)in_sizes; (void)n_in; (void)out_size; (void)ws_size;
}
